// AttentionPooling_9792525435205
// MI455X (gfx1250) — matrix operand order repaired and checked by reading, not compiled, not run on hardware
//
#include <hip/hip_runtime.h>

// CDNA5 / gfx1250 fused attention pooling.
// Score GEMM (x @ W1) in bf16 via v_wmma_f32_16x16x32_bf16, f32 accumulate.
// Each workgroup: 128 contiguous tokens (one spatial row-pair). Each wave owns
// 16 W1 columns and sweeps 8 M-tiles, reusing each B fragment across 8 WMMAs.
// B chunks are double-buffered in LDS and software-pipelined; sched_barriers
// separate (load-issue | compute | pack+store), and sched_group_barriers pin a
// DS8 -> (WMMA1, DS2)x14 -> WMMA2 issue pattern so fragment loads stay ~2 deep
// ahead of the WMMAs (shallow dscnt waits instead of full stalls).

typedef __attribute__((ext_vector_type(16))) __bf16 v16bf;
typedef __attribute__((ext_vector_type(8)))  float  v8f;
typedef __attribute__((ext_vector_type(4)))  unsigned int v4u;

#define DDIM 1024
#define ROWU 516                 // uints per A row (2064 B): 16B aligned, banks stagger by 4
#define AS_U (128 * ROWU)        // 66048 uints (~258 KB)
#define KC   64                  // K-chunk staged per step (2 k-tiles)
#define BWU2 516                 // uints per wave's B region (512 + 4 pad -> bank stagger)
#define BHALF (8 * BWU2)         // one B buffer: 4128 uints (~16 KB)
#define BS_U (2 * BHALF)         // double buffer (~32 KB)

__device__ __forceinline__ unsigned int pack_bf16(float a, float b) {
    unsigned int ua = __builtin_bit_cast(unsigned int, a);
    unsigned int ub = __builtin_bit_cast(unsigned int, b);
    ua = (ua + 0x7FFFu + ((ua >> 16) & 1u)) >> 16;   // RNE
    ub = (ub + 0x7FFFu + ((ub >> 16) & 1u)) >> 16;
    return ua | (ub << 16);
}

__global__ __launch_bounds__(256, 1)
void attnpool_wmma_kernel(const float* __restrict__ x,
                          const float* __restrict__ W1,
                          const float* __restrict__ b1,
                          const float* __restrict__ W2,
                          float* __restrict__ out)
{
    extern __shared__ unsigned int smem[];
    unsigned int* AsU = smem;                         // A, bf16, fragment-permuted
    unsigned int* BsU = smem + AS_U;                  // B double buffer, fragment order
    float*        lg  = (float*)(smem + AS_U + BS_U); // 128 logits

    const int t    = threadIdx.x;
    const int lane = t & 31;
    const int wave = t >> 5;
    const int nn   = lane & 15;
    const int hi   = lane >> 4;

    const int blk = blockIdx.x;
    const int b   = blk >> 5;     // batch
    const int i2  = blk & 31;     // spatial row-pair
    const float* xb = x + ((long)b * 4096 + (long)i2 * 128) * DDIM;

    if (t < 128) lg[t] = 0.f;

    // ---- Stage A: 128x1024 f32 (contiguous 512KB) -> bf16, fragment-permuted ----
    // Per row, per 32-K tile (64B): slots hold K0-7 | K8-15 | K16-23 | K24-31,
    // so each lane's fragment is one contiguous 32B chunk (2 x ds_load_b128).
    for (int it = 0; it < 128; ++it) {
        int e4 = t + it * 256;            // float4 index
        float4 v = ((const float4*)xb)[e4];
        int off = e4 * 4;
        int row = off >> 10;
        int k   = off & 1023;
        int ktl  = k >> 5;
        int g    = (k >> 3) & 3;
        int slot = g;                        // identity {0,1,2,3}
        int ui = row * ROWU + ktl * 16 + slot * 4 + ((k & 7) >> 1);
        AsU[ui]     = pack_bf16(v.x, v.y);
        AsU[ui + 1] = pack_bf16(v.z, v.w);
    }

    // Per-thread staging coordinates (constant across chunks/rounds).
    const int s_cg  = t & 31;              // column group (4 cols)
    const int s_w   = (s_cg * 4) >> 4;     // owning wave of those columns
    const int s_ncb = (s_cg * 4) & 15;

    const v8f vzero = {0.f, 0.f, 0.f, 0.f, 0.f, 0.f, 0.f, 0.f};

    for (int r = 0; r < 8; ++r) {          // N rounds: 128 W1 columns each
        v8f acc[8];
        #pragma unroll
        for (int q = 0; q < 8; ++q) acc[q] = vzero;

        // ---- prologue: stage chunk 0 into buffer 0 ----
        #pragma unroll
        for (int i = 0; i < 4; ++i) {
            int k2 = (t >> 5) + 8 * i;     // uint-K index 0..31
            const float* w1p = W1 + (long)(2 * k2) * DDIM + r * 128 + s_cg * 4;
            float4 r0 = *(const float4*)(w1p);
            float4 r1 = *(const float4*)(w1p + DDIM);
            int bhi = (k2 >> 3) & 1, ktl = k2 >> 4, klo2 = k2 & 7;
            unsigned int* dst = &BsU[s_w * BWU2 + ktl * 256 + (s_ncb + 16 * bhi) * 8 + klo2];
            dst[0]  = pack_bf16(r0.x, r1.x);
            dst[8]  = pack_bf16(r0.y, r1.y);
            dst[16] = pack_bf16(r0.z, r1.z);
            dst[24] = pack_bf16(r0.w, r1.w);
        }
        __syncthreads();

        for (int c = 0; c < 16; ++c) {     // K chunks of 64
            const int p = c & 1;
            // ---- phase 1: issue global loads for chunk c+1 ----
            float4 L0[4], L1[4];
            if (c < 15) {
                const int kc0 = (c + 1) * KC;
                #pragma unroll
                for (int i = 0; i < 4; ++i) {
                    int k2 = (t >> 5) + 8 * i;
                    const float* w1p = W1 + (long)(kc0 + 2 * k2) * DDIM + r * 128 + s_cg * 4;
                    L0[i] = *(const float4*)(w1p);
                    L1[i] = *(const float4*)(w1p + DDIM);
                }
            }
            __builtin_amdgcn_sched_barrier(0);  // keep load-issue above the WMMA block
            // ---- phase 2: compute chunk c (ds_load_b128 + WMMA only) ----
            #pragma unroll
            for (int kt = 0; kt < 2; ++kt) {
                union { v16bf v; v4u q[2]; } bfr;
                const v4u* bp = (const v4u*)&BsU[p * BHALF + wave * BWU2 + kt * 256 + lane * 8];
                bfr.q[0] = bp[0];
                bfr.q[1] = bp[1];
                const int ktile = c * 2 + kt;
                #pragma unroll
                for (int m = 0; m < 8; ++m) {
                    union { v16bf v; v4u q[2]; } af;
                    const v4u* ap = (const v4u*)&AsU[(m * 16 + nn) * ROWU + ktile * 16 + hi * 8];
                    af.q[0] = ap[0];
                    af.q[1] = ap[1];
                    acc[m] = __builtin_amdgcn_wmma_f32_16x16x32_bf16(
                        false, af.v, false, bfr.v, (short)0, acc[m], false, false);
                }
            }
            // Pin issue order in this region: 8 DS reads up front (B frags + first
            // A frags), then each WMMA trailed by 2 more DS reads (next A frag).
            // 36 DS reads, 16 WMMAs total per chunk.
            __builtin_amdgcn_sched_group_barrier(0x100, 8, 0);   // DS read x8
            #pragma unroll
            for (int gi = 0; gi < 14; ++gi) {
                __builtin_amdgcn_sched_group_barrier(0x008, 1, 0); // WMMA x1
                __builtin_amdgcn_sched_group_barrier(0x100, 2, 0); // DS read x2
            }
            __builtin_amdgcn_sched_group_barrier(0x008, 2, 0);   // WMMA x2
            __builtin_amdgcn_sched_barrier(0);  // keep pack+store below the WMMA block
            // ---- phase 3: pack + store chunk c+1 into the other buffer ----
            if (c < 15) {
                #pragma unroll
                for (int i = 0; i < 4; ++i) {
                    int k2 = (t >> 5) + 8 * i;
                    int bhi = (k2 >> 3) & 1, ktl = k2 >> 4, klo2 = k2 & 7;
                    unsigned int* dst = &BsU[(p ^ 1) * BHALF + s_w * BWU2 + ktl * 256 +
                                             (s_ncb + 16 * bhi) * 8 + klo2];
                    dst[0]  = pack_bf16(L0[i].x, L1[i].x);
                    dst[8]  = pack_bf16(L0[i].y, L1[i].y);
                    dst[16] = pack_bf16(L0[i].z, L1[i].z);
                    dst[24] = pack_bf16(L0[i].w, L1[i].w);
                }
            }
            __syncthreads();
        }
        // ---- Fold this round's 16 columns into logits: +b1, ReLU, *W2, reduce ----
        const int ncol = r * 128 + wave * 16 + nn;
        const float b1n = b1[ncol];
        const float w2n = W2[ncol];
        #pragma unroll
        for (int m = 0; m < 8; ++m) {
            #pragma unroll
            for (int v = 0; v < 8; ++v) {
                float h = acc[m][v] + b1n;
                h = fmaxf(h, 0.f) * w2n;
                h += __shfl_xor(h, 1);
                h += __shfl_xor(h, 2);
                h += __shfl_xor(h, 4);
                h += __shfl_xor(h, 8);
                if (nn == 0) atomicAdd(&lg[m * 16 + hi * 8 + v], h);
            }
        }
    }
    __syncthreads();

    // ---- Softmax over each 2x2 window + fp32 weighted sum of x ----
    // window j: tokens {2j, 2j+1, 64+2j, 65+2j}; region r = i2*32 + j
    float* ob = out + ((long)b * 1024 + (long)i2 * 32) * DDIM;
    for (int q = 0; q < 4; ++q) {
        int j = wave + q * 8;
        float l0 = lg[2 * j],      l1 = lg[2 * j + 1];
        float l2 = lg[64 + 2 * j], l3 = lg[65 + 2 * j];
        float mx = fmaxf(fmaxf(l0, l1), fmaxf(l2, l3));
        float e0 = __expf(l0 - mx), e1 = __expf(l1 - mx);
        float e2 = __expf(l2 - mx), e3 = __expf(l3 - mx);
        float inv = 1.f / (e0 + e1 + e2 + e3);
        e0 *= inv; e1 *= inv; e2 *= inv; e3 *= inv;
        const float4* xa = (const float4*)(xb + (long)(2 * j) * DDIM);
        const float4* xc = (const float4*)(xb + (long)(2 * j + 1) * DDIM);
        const float4* xe = (const float4*)(xb + (long)(64 + 2 * j) * DDIM);
        const float4* xg = (const float4*)(xb + (long)(65 + 2 * j) * DDIM);
        float4* o = (float4*)(ob + (long)j * DDIM);
        for (int it = lane; it < DDIM / 4; it += 32) {
            float4 a = xa[it], c = xc[it], e = xe[it], g = xg[it];
            float4 rr;
            rr.x = e0 * a.x + e1 * c.x + e2 * e.x + e3 * g.x;
            rr.y = e0 * a.y + e1 * c.y + e2 * e.y + e3 * g.y;
            rr.z = e0 * a.z + e1 * c.z + e2 * e.z + e3 * g.z;
            rr.w = e0 * a.w + e1 * c.w + e2 * e.w + e3 * g.w;
            o[it] = rr;
        }
    }
}

extern "C" void kernel_launch(void* const* d_in, const int* in_sizes, int n_in,
                              void* d_out, int out_size, void* d_ws, size_t ws_size,
                              hipStream_t stream) {
    (void)n_in; (void)out_size; (void)d_ws; (void)ws_size;
    const float* x  = (const float*)d_in[0];
    const float* W1 = (const float*)d_in[1];
    const float* b1 = (const float*)d_in[2];
    const float* W2 = (const float*)d_in[3];
    // b2 (d_in[4]) is a uniform logit shift; softmax cancels it.
    float* out = (float*)d_out;

    const int B = in_sizes[0] / (4096 * 1024);   // 16
    const int blocks = B * 32;                   // one row-pair per workgroup
    const size_t shmem = (size_t)(AS_U + BS_U) * 4 + 128 * 4;  // ~298 KB of 320 KB LDS

    attnpool_wmma_kernel<<<blocks, 256, shmem, stream>>>(x, W1, b1, W2, out);
}
